// DeepLocModel_2894807958254
// MI455X (gfx1250) — compile-verified, matrix operand order checked by reading, not run on hardware
//
#include <hip/hip_runtime.h>
#include <cstddef>

#define BSZ   128
#define LSEQ  1000
#define EDIM  32
#define NFTOT 120
#define C2    128
#define HENC  256
#define HDEC  512

typedef __attribute__((ext_vector_type(16))) __bf16         v16bf;
typedef __attribute__((ext_vector_type(8)))  float          v8f;
typedef __attribute__((ext_vector_type(8)))  unsigned short us8;
typedef __attribute__((ext_vector_type(16))) unsigned short us16;

__device__ __forceinline__ unsigned short f2bf(float f) {
  union { float f; unsigned u; } x; x.f = f;
  unsigned r = x.u + 0x7FFFu + ((x.u >> 16) & 1u);   // round-to-nearest-even
  return (unsigned short)(r >> 16);
}
__device__ __forceinline__ float bf2f(unsigned short h) {
  union { unsigned u; float f; } x; x.u = ((unsigned)h) << 16;
  return x.f;
}
__device__ __forceinline__ float sigm(float x) { return 1.0f / (1.0f + __expf(-x)); }

// Load one lane's 16-bf16 WMMA fragment (A or B operand).
// ISA layout (16-bit A 16x32): elem e<8 -> k = k0 + 8*half + e
//                              elem e>=8 -> k = k0 + 16 + 8*half + (e-8)
// i.e. two contiguous 8-element (16B) runs -> two b128 loads.
__device__ __forceinline__ v16bf load_frag(const unsigned short* row, int k0, int half) {
  us8 lo = *(const us8*)(row + k0 + 8 * half);
  us8 hi = *(const us8*)(row + k0 + 8 * half + 16);
  us16 t;
#pragma unroll
  for (int i = 0; i < 8; ++i) { t[i] = lo[i]; t[8 + i] = hi[i]; }
  return __builtin_bit_cast(v16bf, t);
}

__device__ __forceinline__ v8f wmma_bf16(v16bf a, v16bf b, v8f c) {
  return __builtin_amdgcn_wmma_f32_16x16x32_bf16(false, a, false, b, (short)0, c, false, false);
}

// ---------------------------------------------------------------- utilities
__global__ void f2bf_kernel(const float* __restrict__ s, unsigned short* __restrict__ d, int n) {
  int i = blockIdx.x * 256 + threadIdx.x;
  if (i < n) d[i] = f2bf(s[i]);
}

// zero the grid-barrier counters + the 256-entry zero row (h_{-1} == 0 trick)
__global__ void init_sync_kernel(unsigned* bars, unsigned short* zrow) {
  int t = threadIdx.x;
  if (t < 128) bars[t] = 0u;
  zrow[t] = 0;
}

// ---------------------------------------------------------------- embedding
// xe layout: [B][E][L] f32
__global__ void embed_kernel(const int* __restrict__ x, const float* __restrict__ ew,
                             float* __restrict__ xe) {
  int i = blockIdx.x * 256 + threadIdx.x;
  if (i >= BSZ * LSEQ) return;
  int b = i / LSEQ, l = i % LSEQ;
  const float* row = ew + x[i] * EDIM;
#pragma unroll
  for (int e = 0; e < EDIM; ++e) xe[((size_t)b * EDIM + e) * LSEQ + l] = row[e];
}

// ---------------------------------------------------------------- conv branches
#define LT1 40
__global__ void __launch_bounds__(128) conv1_kernel(
    const float* __restrict__ xe,
    const float* __restrict__ w0, const float* __restrict__ b0,
    const float* __restrict__ w1, const float* __restrict__ b1,
    const float* __restrict__ w2, const float* __restrict__ b2,
    const float* __restrict__ w3, const float* __restrict__ b3,
    const float* __restrict__ w4, const float* __restrict__ b4,
    const float* __restrict__ w5, const float* __restrict__ b5,
    float* __restrict__ cnn) {
  __shared__ float xs[EDIM][LT1 + 20];
  const int blk = blockIdx.x;
  const int b = blk / (LSEQ / LT1);
  const int l0 = (blk % (LSEQ / LT1)) * LT1;
  for (int idx = threadIdx.x; idx < EDIM * (LT1 + 20); idx += 128) {
    int ci = idx / (LT1 + 20), c = idx % (LT1 + 20);
    int l = l0 + c - 10;
    xs[ci][c] = (l >= 0 && l < LSEQ) ? xe[((size_t)b * EDIM + ci) * LSEQ + l] : 0.f;
  }
  __syncthreads();
  const int co = threadIdx.x;
  if (co >= NFTOT) return;
  const int Ks[6] = {1, 3, 5, 9, 15, 21};
  const float* ws[6] = {w0, w1, w2, w3, w4, w5};
  const float* bs[6] = {b0, b1, b2, b3, b4, b5};
  const int br = co / 20, cc = co % 20;
  const int K = Ks[br], p = K / 2;
  const float* w = ws[br] + (size_t)cc * EDIM * K;
  const float bias = bs[br][cc];
  for (int l = 0; l < LT1; ++l) {
    float a = bias;
    for (int ci = 0; ci < EDIM; ++ci) {
      const float* wr = w + ci * K;
      const int base = l + 10 - p;
      for (int k = 0; k < K; ++k) a += wr[k] * xs[ci][base + k];
    }
    cnn[((size_t)b * NFTOT + co) * LSEQ + (l0 + l)] = fmaxf(a, 0.f);
  }
}

#define LT2 50
// writes featbf layout [L][B][128] bf16
__global__ void __launch_bounds__(128) conv2_kernel(
    const float* __restrict__ cnn, const float* __restrict__ w,
    const float* __restrict__ bias, unsigned short* __restrict__ featbf) {
  __shared__ float cs[NFTOT][LT2 + 2];
  const int blk = blockIdx.x;
  const int b = blk / (LSEQ / LT2);
  const int l0 = (blk % (LSEQ / LT2)) * LT2;
  for (int idx = threadIdx.x; idx < NFTOT * (LT2 + 2); idx += 128) {
    int c = idx / (LT2 + 2), cc = idx % (LT2 + 2);
    int l = l0 + cc - 1;
    cs[c][cc] = (l >= 0 && l < LSEQ) ? cnn[((size_t)b * NFTOT + c) * LSEQ + l] : 0.f;
  }
  __syncthreads();
  const int co = threadIdx.x;
  const float* wr = w + (size_t)co * NFTOT * 3;
  const float bv = bias[co];
  for (int l = 0; l < LT2; ++l) {
    float a = bv;
    for (int c = 0; c < NFTOT; ++c)
      a += wr[c * 3 + 0] * cs[c][l] + wr[c * 3 + 1] * cs[c][l + 1] + wr[c * 3 + 2] * cs[c][l + 2];
    featbf[(((size_t)(l0 + l)) * BSZ + b) * C2 + co] = f2bf(fmaxf(a, 0.f));
  }
}

// ---------------------------------------------------------------- BiLSTM (persistent)
// grid = 32 blocks x 256 threads. dir = blk>>4. WG wg owns hidden units u0..u0+15
// with all 4 gate rows (i,f,g,o) for that slice -> 4*16 rows x 384 K in LDS (48KB),
// resident for all 1000 steps. Cell state in registers. h via L2 + grid barrier
// (or HW cluster barrier when dispatched as one 16-WG cluster per direction).
// k-loop is software-pipelined: next chunk's A (global) + 4 B (LDS) fragments are
// in flight while the current chunk's four WMMAs execute.
__global__ void __launch_bounds__(256) lstm_kernel(
    const unsigned short* __restrict__ featbf,
    const unsigned short* __restrict__ wifb, const unsigned short* __restrict__ whfb,
    const unsigned short* __restrict__ wirb, const unsigned short* __restrict__ whrb,
    const float* __restrict__ bif, const float* __restrict__ bhf,
    const float* __restrict__ bir, const float* __restrict__ bhr,
    unsigned short* __restrict__ hsf, unsigned short* __restrict__ hsb,
    const unsigned short* __restrict__ zrow, unsigned* bars, int use_cluster) {
  __shared__ __align__(16) unsigned short Wlds[4 * 16 * 384];
  __shared__ float biasLds[64];
  const int dir = blockIdx.x >> 4;
  const int wg = blockIdx.x & 15;
  const int u0 = wg * 16;
  const int tid = threadIdx.x;
  const unsigned short* wih = dir ? wirb : wifb;
  const unsigned short* whh = dir ? whrb : whfb;
  const float* bi = dir ? bir : bif;
  const float* bh = dir ? bhr : bhf;
  unsigned short* hs = dir ? hsb : hsf;
  unsigned* bar = bars + dir * 64;

  for (int idx = tid; idx < 4 * 16 * 384; idx += 256) {
    const int k = idx % 384;
    const int rn = idx / 384;
    const int j = (rn >> 4) * HENC + u0 + (rn & 15);
    Wlds[idx] = (k < C2) ? wih[(size_t)j * C2 + k] : whh[(size_t)j * HENC + (k - C2)];
  }
  if (tid < 64) {
    const int j = (tid >> 4) * HENC + u0 + (tid & 15);
    biasLds[tid] = bi[j] + bh[j];
  }
  __syncthreads();

  const int wave = tid >> 5;
  const int lane = tid & 31;
  const int half = lane >> 4, mn = lane & 15;
  const int bb = wave * 16 + mn;

  float bias_r[4];  // hoist gate biases out of the scan (compiler can't cross barriers)
#pragma unroll
  for (int g = 0; g < 4; ++g) bias_r[g] = biasLds[g * 16 + mn];

  v8f creg;
#pragma unroll
  for (int v = 0; v < 8; ++v) creg[v] = 0.f;

  unsigned target = 0;
  for (int s = 0; s < LSEQ; ++s) {
    const int t = dir ? (LSEQ - 1 - s) : s;
    const unsigned short* xrow = featbf + ((size_t)t * BSZ + bb) * C2;
    // h_{-1} reads come from a zeroed row -> identical pipelined loop for every step
    const unsigned short* hrow =
        (s > 0) ? hs + ((size_t)(dir ? t + 1 : t - 1) * BSZ + bb) * HENC : zrow;
    if (s + 1 < LSEQ) {
      const int tn = dir ? t - 1 : t + 1;
      __builtin_prefetch(featbf + ((size_t)tn * BSZ + bb) * C2, 0, 1);
    }
    v8f acc[4];
#pragma unroll
    for (int g = 0; g < 4; ++g) {
#pragma unroll
      for (int v = 0; v < 8; ++v) acc[g][v] = bias_r[g];
    }
    // software-pipelined GEMM over K = 384 (12 chunks of 32)
    v16bf a_cur = load_frag(xrow, 0, half);
    v16bf b_cur[4];
#pragma unroll
    for (int g = 0; g < 4; ++g) b_cur[g] = load_frag(&Wlds[(size_t)(g * 16 + mn) * 384], 0, half);
#pragma unroll
    for (int kc = 0; kc < 12; ++kc) {
      v16bf a_nxt = a_cur;
      v16bf b_nxt[4] = {b_cur[0], b_cur[1], b_cur[2], b_cur[3]};
      if (kc + 1 < 12) {
        const int kn = (kc + 1) * 32;
        a_nxt = (kn < C2) ? load_frag(xrow, kn, half) : load_frag(hrow, kn - C2, half);
#pragma unroll
        for (int g = 0; g < 4; ++g)
          b_nxt[g] = load_frag(&Wlds[(size_t)(g * 16 + mn) * 384], kn, half);
      }
#pragma unroll
      for (int g = 0; g < 4; ++g) acc[g] = wmma_bf16(a_cur, b_cur[g], acc[g]);
      a_cur = a_nxt;
#pragma unroll
      for (int g = 0; g < 4; ++g) b_cur[g] = b_nxt[g];
    }
#pragma unroll
    for (int v = 0; v < 8; ++v) {
      const float iv = sigm(acc[0][v]);
      const float fv = sigm(acc[1][v]);
      const float gv = tanhf(acc[2][v]);
      const float ov = sigm(acc[3][v]);
      const float c = fv * creg[v] + iv * gv;
      creg[v] = c;
      const int brow = wave * 16 + v + 8 * half;
      hs[((size_t)t * BSZ + brow) * HENC + (u0 + mn)] = f2bf(ov * tanhf(c));
    }
    __syncthreads();
    target += 16;
    if (use_cluster) {
      // CDNA5 fast path: one 16-WG cluster per direction, HW split barrier.
      __builtin_amdgcn_fence(__ATOMIC_RELEASE, "agent");
      if (tid < 32) asm volatile("s_barrier_signal -3" ::: "memory");
      asm volatile("s_barrier_wait -3" ::: "memory");
      __builtin_amdgcn_fence(__ATOMIC_ACQUIRE, "agent");
    } else {
      if (tid == 0) {
        __hip_atomic_fetch_add(bar, 1u, __ATOMIC_RELEASE, __HIP_MEMORY_SCOPE_AGENT);
        while (__hip_atomic_load(bar, __ATOMIC_ACQUIRE, __HIP_MEMORY_SCOPE_AGENT) < target)
          __builtin_amdgcn_s_sleep(2);
      }
    }
    __syncthreads();
  }
}

// ---------------------------------------------------------------- e_enc = lstm_out @ We^T + be
// M = L*B rows (r = l*128 + b), K = 512 ([hsf|hsb]), N = 256. bf16 out [L][B][256].
__global__ void __launch_bounds__(256) eenc_kernel(
    const unsigned short* __restrict__ hsf, const unsigned short* __restrict__ hsb,
    const unsigned short* __restrict__ wewb, const float* __restrict__ web,
    unsigned short* __restrict__ eenc) {
  const int rt = blockIdx.x * 8 + (threadIdx.x >> 5);  // 0..7999
  const int lane = threadIdx.x & 31;
  const int half = lane >> 4, mn = lane & 15;
  const size_t r = (size_t)rt * 16 + mn;
  const unsigned short* arf = hsf + r * HENC;
  const unsigned short* arb = hsb + r * HENC;
  for (int nt = 0; nt < 16; ++nt) {
    const int n = nt * 16 + mn;
    const unsigned short* brow = wewb + (size_t)n * 512;
    v8f acc;
    const float bsv = web[n];
#pragma unroll
    for (int v = 0; v < 8; ++v) acc[v] = bsv;
#pragma unroll
    for (int kc = 0; kc < 16; ++kc) {
      const int k0 = kc * 32;
      v16bf a = (k0 < HENC) ? load_frag(arf, k0, half) : load_frag(arb, k0 - HENC, half);
      v16bf bfr = load_frag(brow, k0, half);
      acc = wmma_bf16(a, bfr, acc);
    }
#pragma unroll
    for (int v = 0; v < 8; ++v) {
      const size_t rr = (size_t)rt * 16 + v + 8 * half;
      eenc[rr * 256 + n] = f2bf(acc[v]);
    }
  }
}

// ---------------------------------------------------------------- decoder
__global__ void dec_init_kernel(const float* __restrict__ c0, unsigned short* __restrict__ ctxbf,
                                float* __restrict__ ctxf, unsigned short* __restrict__ hd,
                                float* __restrict__ cd) {
  int i = blockIdx.x * 256 + threadIdx.x;
  if (i < BSZ * HDEC) {
    float v = c0[i & (HDEC - 1)];
    ctxf[i] = v;
    ctxbf[i] = f2bf(v);
    hd[i] = 0;
    cd[i] = 0.f;
  }
}

// gates = [ctx|hd] @ Wi^T + hd @ Wh^T + b  -> LSTMCell update.
// K = 1536 packed as ctx(512)|hd(512)|hd(512). N = 2048 = 4 gates x 512.
// B streams from global (L2-resident 6MB) -> software-pipelined double buffer.
__global__ void __launch_bounds__(256) dec_gate_kernel(
    const unsigned short* __restrict__ ctxbf, const unsigned short* __restrict__ hdin,
    const unsigned short* __restrict__ dwib, const unsigned short* __restrict__ dwhb,
    const float* __restrict__ dbi, const float* __restrict__ dbh, float* __restrict__ cd,
    unsigned short* __restrict__ hdout) {
  const int W = blockIdx.x * 8 + (threadIdx.x >> 5);  // 0..255
  const int mtile = W >> 5;                           // 0..7
  const int u0 = (W & 31) * 16;                       // 0..496
  const int lane = threadIdx.x & 31;
  const int half = lane >> 4, mn = lane & 15;
  const int bb = mtile * 16 + mn;
  const unsigned short* actx = ctxbf + (size_t)bb * HDEC;
  const unsigned short* ahd = hdin + (size_t)bb * HDEC;

  v8f acc[4];
  int jrow[4];
#pragma unroll
  for (int g = 0; g < 4; ++g) {
    const int j = g * HDEC + u0 + mn;
    jrow[g] = j;
    const float bsv = dbi[j] + dbh[j];
#pragma unroll
    for (int v = 0; v < 8; ++v) acc[g][v] = bsv;
  }
  auto loadA = [&](int k0) -> v16bf {
    if (k0 < 512) return load_frag(actx, k0, half);
    if (k0 < 1024) return load_frag(ahd, k0 - 512, half);
    return load_frag(ahd, k0 - 1024, half);
  };
  auto loadB = [&](int g, int k0) -> v16bf {
    if (k0 < 1024) return load_frag(dwib + (size_t)jrow[g] * 1024, k0, half);
    return load_frag(dwhb + (size_t)jrow[g] * 512, k0 - 1024, half);
  };
  v16bf a_cur = loadA(0);
  v16bf b_cur[4];
#pragma unroll
  for (int g = 0; g < 4; ++g) b_cur[g] = loadB(g, 0);
#pragma unroll
  for (int kc = 0; kc < 48; ++kc) {
    v16bf a_nxt = a_cur;
    v16bf b_nxt[4] = {b_cur[0], b_cur[1], b_cur[2], b_cur[3]};
    if (kc + 1 < 48) {
      const int kn = (kc + 1) * 32;
      a_nxt = loadA(kn);
#pragma unroll
      for (int g = 0; g < 4; ++g) b_nxt[g] = loadB(g, kn);
    }
#pragma unroll
    for (int g = 0; g < 4; ++g) acc[g] = wmma_bf16(a_cur, b_cur[g], acc[g]);
    a_cur = a_nxt;
#pragma unroll
    for (int g = 0; g < 4; ++g) b_cur[g] = b_nxt[g];
  }
#pragma unroll
  for (int v = 0; v < 8; ++v) {
    const int brow = mtile * 16 + v + 8 * half;
    const int u = u0 + mn;
    const float iv = sigm(acc[0][v]);
    const float fv = sigm(acc[1][v]);
    const float gv = tanhf(acc[2][v]);
    const float ov = sigm(acc[3][v]);
    const float c = fv * cd[(size_t)brow * HDEC + u] + iv * gv;
    cd[(size_t)brow * HDEC + u] = c;
    hdout[(size_t)brow * HDEC + u] = f2bf(ov * tanhf(c));
  }
}

// d = hd @ Wd^T + bd   [128 x 256], K=512
__global__ void __launch_bounds__(256) dec_d_kernel(
    const unsigned short* __restrict__ hd, const unsigned short* __restrict__ wdwb,
    const float* __restrict__ wdb, float* __restrict__ dbuf) {
  const int W = blockIdx.x * 8 + (threadIdx.x >> 5);  // 0..127
  const int mtile = W >> 4;
  const int ntile = W & 15;
  const int lane = threadIdx.x & 31;
  const int half = lane >> 4, mn = lane & 15;
  const int bb = mtile * 16 + mn;
  const int n = ntile * 16 + mn;
  const unsigned short* arow = hd + (size_t)bb * HDEC;
  const unsigned short* brow = wdwb + (size_t)n * HDEC;
  v8f acc;
  const float bsv = wdb[n];
#pragma unroll
  for (int v = 0; v < 8; ++v) acc[v] = bsv;
#pragma unroll
  for (int kc = 0; kc < 16; ++kc) {
    v16bf a = load_frag(arow, kc * 32, half);
    v16bf b = load_frag(brow, kc * 32, half);
    acc = wmma_bf16(a, b, acc);
  }
#pragma unroll
  for (int v = 0; v < 8; ++v) {
    const int brw = mtile * 16 + v + 8 * half;
    dbuf[(size_t)brw * 256 + n] = acc[v];
  }
}

// a[b,l] = attn_v . tanh(e_enc[l,b,:] + d[b,:])  -- one wave per (b,l)
__global__ void __launch_bounds__(256) attn_score_kernel(
    const unsigned short* __restrict__ eenc, const float* __restrict__ dbuf,
    const float* __restrict__ attnv, float* __restrict__ abuf) {
  const int W = blockIdx.x * 8 + (threadIdx.x >> 5);  // 0..127999
  const int b = W & (BSZ - 1);
  const int l = W >> 7;
  const int lane = threadIdx.x & 31;
  const unsigned short* erow = eenc + ((size_t)l * BSZ + b) * 256 + lane * 8;
  const float* drow = dbuf + (size_t)b * 256 + lane * 8;
  float s = 0.f;
#pragma unroll
  for (int i = 0; i < 8; ++i) s += attnv[lane * 8 + i] * tanhf(bf2f(erow[i]) + drow[i]);
#pragma unroll
  for (int off = 16; off; off >>= 1) s += __shfl_xor(s, off, 32);
  if (lane == 0) abuf[(size_t)b * LSEQ + l] = s;
}

// softmax over l + ctx[b,:] = sum_l p_l * lstm_out[b,l,:]   -- one block per b
__global__ void __launch_bounds__(256) attn_ctx_kernel(
    float* __restrict__ abuf, const unsigned short* __restrict__ hsf,
    const unsigned short* __restrict__ hsb, float* __restrict__ ctxf,
    unsigned short* __restrict__ ctxbf) {
  __shared__ float red[256];
  const int b = blockIdx.x;
  const int tid = threadIdx.x;
  float m = -1e30f;
  for (int l = tid; l < LSEQ; l += 256) m = fmaxf(m, abuf[(size_t)b * LSEQ + l]);
  red[tid] = m;
  __syncthreads();
  for (int s2 = 128; s2; s2 >>= 1) {
    if (tid < s2) red[tid] = fmaxf(red[tid], red[tid + s2]);
    __syncthreads();
  }
  const float mx = red[0];
  __syncthreads();
  float sum = 0.f;
  for (int l = tid; l < LSEQ; l += 256) {
    float e = __expf(abuf[(size_t)b * LSEQ + l] - mx);
    abuf[(size_t)b * LSEQ + l] = e;
    sum += e;
  }
  red[tid] = sum;
  __syncthreads();
  for (int s2 = 128; s2; s2 >>= 1) {
    if (tid < s2) red[tid] += red[tid + s2];
    __syncthreads();
  }
  const float inv = 1.0f / red[0];
  __syncthreads();
  float a0 = 0.f, a1 = 0.f;
  for (int l = 0; l < LSEQ; ++l) {
    const float p = abuf[(size_t)b * LSEQ + l];
    const size_t ro = ((size_t)l * BSZ + b) * HENC + tid;
    a0 += p * bf2f(hsf[ro]);
    a1 += p * bf2f(hsb[ro]);
  }
  a0 *= inv;
  a1 *= inv;
  ctxf[(size_t)b * HDEC + tid] = a0;
  ctxf[(size_t)b * HDEC + 256 + tid] = a1;
  ctxbf[(size_t)b * HDEC + tid] = f2bf(a0);
  ctxbf[(size_t)b * HDEC + 256 + tid] = f2bf(a1);
}

// z = relu(ctx @ fc^T + b); out = sigmoid(z @ ow^T + ob)  -- one block per b
__global__ void __launch_bounds__(256) head_kernel(
    const float* __restrict__ ctxf, const float* __restrict__ fcw, const float* __restrict__ fcb,
    const float* __restrict__ ow, const float* __restrict__ ob, float* __restrict__ out) {
  __shared__ float z[512];
  const int b = blockIdx.x;
  const int tid = threadIdx.x;
  const float* c = ctxf + (size_t)b * HDEC;
  for (int j = tid; j < 512; j += 256) {
    float a = fcb[j];
    const float* w = fcw + (size_t)j * 512;
    for (int k = 0; k < 512; ++k) a += w[k] * c[k];
    z[j] = fmaxf(a, 0.f);
  }
  __syncthreads();
  if (tid < 8) {
    float a = ob[tid];
    const float* w = ow + (size_t)tid * 512;
    for (int k = 0; k < 512; ++k) a += w[k] * z[k];
    out[b * 8 + tid] = 1.0f / (1.0f + __expf(-a));
  }
}

// ================================================================ host
extern "C" void kernel_launch(void* const* d_in, const int* in_sizes, int n_in, void* d_out,
                              int out_size, void* d_ws, size_t ws_size, hipStream_t stream) {
  (void)in_sizes; (void)n_in; (void)out_size; (void)ws_size;
  const int*   x    = (const int*)d_in[0];
  const float* emw  = (const float*)d_in[1];
  const float* cw0 = (const float*)d_in[2],  *cb0 = (const float*)d_in[3];
  const float* cw1 = (const float*)d_in[4],  *cb1 = (const float*)d_in[5];
  const float* cw2 = (const float*)d_in[6],  *cb2 = (const float*)d_in[7];
  const float* cw3 = (const float*)d_in[8],  *cb3 = (const float*)d_in[9];
  const float* cw4 = (const float*)d_in[10], *cb4 = (const float*)d_in[11];
  const float* cw5 = (const float*)d_in[12], *cb5 = (const float*)d_in[13];
  const float* c2w = (const float*)d_in[14], *c2b = (const float*)d_in[15];
  const float* wif = (const float*)d_in[16], *whf = (const float*)d_in[17];
  const float* bif = (const float*)d_in[18], *bhf = (const float*)d_in[19];
  const float* wir = (const float*)d_in[20], *whr = (const float*)d_in[21];
  const float* bir = (const float*)d_in[22], *bhr = (const float*)d_in[23];
  const float* dwi = (const float*)d_in[24], *dwh = (const float*)d_in[25];
  const float* dbi = (const float*)d_in[26], *dbh = (const float*)d_in[27];
  const float* attnv = (const float*)d_in[28];
  const float* wew = (const float*)d_in[29], *web = (const float*)d_in[30];
  const float* wdw = (const float*)d_in[31], *wdb = (const float*)d_in[32];
  const float* fcw = (const float*)d_in[33], *fcb = (const float*)d_in[34];
  const float* ow  = (const float*)d_in[35], *ob  = (const float*)d_in[36];
  const float* c0  = (const float*)d_in[37];

  char* base = (char*)d_ws;
  size_t off = 0;
  auto alloc = [&](size_t bytes) -> char* {
    off = (off + 255) & ~(size_t)255;
    char* p = base + off;
    off += bytes;
    return p;
  };

  unsigned short* featbf = (unsigned short*)alloc((size_t)LSEQ * BSZ * C2 * 2);
  unsigned short* hsf    = (unsigned short*)alloc((size_t)LSEQ * BSZ * HENC * 2);
  unsigned short* hsb    = (unsigned short*)alloc((size_t)LSEQ * BSZ * HENC * 2);
  unsigned short* wifb = (unsigned short*)alloc((size_t)1024 * 128 * 2);
  unsigned short* whfb = (unsigned short*)alloc((size_t)1024 * 256 * 2);
  unsigned short* wirb = (unsigned short*)alloc((size_t)1024 * 128 * 2);
  unsigned short* whrb = (unsigned short*)alloc((size_t)1024 * 256 * 2);
  unsigned short* wewbf = (unsigned short*)alloc((size_t)256 * 512 * 2);
  unsigned short* dwib = (unsigned short*)alloc((size_t)2048 * 1024 * 2);
  unsigned short* dwhb = (unsigned short*)alloc((size_t)2048 * 512 * 2);
  unsigned short* wdwb = (unsigned short*)alloc((size_t)256 * 512 * 2);
  unsigned short* ctxbf = (unsigned short*)alloc((size_t)BSZ * HDEC * 2);
  float* ctxf = (float*)alloc((size_t)BSZ * HDEC * 4);
  unsigned short* hdA = (unsigned short*)alloc((size_t)BSZ * HDEC * 2);
  unsigned short* hdB = (unsigned short*)alloc((size_t)BSZ * HDEC * 2);
  float* cd   = (float*)alloc((size_t)BSZ * HDEC * 4);
  float* dbuf = (float*)alloc((size_t)BSZ * 256 * 4);
  float* abuf = (float*)alloc((size_t)BSZ * LSEQ * 4);
  unsigned* bars = (unsigned*)alloc(512);
  unsigned short* zrow = (unsigned short*)alloc(256 * 2);  // zeroed h_{-1} row
  // xe + cnn are dead once featbf exists; eenc aliases that region.
  char* region = alloc((size_t)BSZ * EDIM * LSEQ * 4 + (size_t)BSZ * NFTOT * LSEQ * 4);
  float* xe  = (float*)region;
  float* cnn = (float*)(region + (size_t)BSZ * EDIM * LSEQ * 4);
  unsigned short* eenc = (unsigned short*)region;  // [L][B][256] bf16 (65.5MB < region)

  auto cvt = [&](const float* s, unsigned short* d, int n) {
    f2bf_kernel<<<(n + 255) / 256, 256, 0, stream>>>(s, d, n);
  };
  cvt(wif, wifb, 1024 * 128);
  cvt(whf, whfb, 1024 * 256);
  cvt(wir, wirb, 1024 * 128);
  cvt(whr, whrb, 1024 * 256);
  cvt(wew, wewbf, 256 * 512);
  cvt(dwi, dwib, 2048 * 1024);
  cvt(dwh, dwhb, 2048 * 512);
  cvt(wdw, wdwb, 256 * 512);

  embed_kernel<<<(BSZ * LSEQ + 255) / 256, 256, 0, stream>>>(x, emw, xe);
  conv1_kernel<<<BSZ * (LSEQ / LT1), 128, 0, stream>>>(xe, cw0, cb0, cw1, cb1, cw2, cb2, cw3, cb3,
                                                       cw4, cb4, cw5, cb5, cnn);
  conv2_kernel<<<BSZ * (LSEQ / LT2), 128, 0, stream>>>(cnn, c2w, c2b, featbf);

  init_sync_kernel<<<1, 256, 0, stream>>>(bars, zrow);
  // use_cluster=0: plain dispatch (no cluster launch API) -> L2 atomic barrier path.
  // If dispatched as one 16-WG cluster per direction, pass 1 to use the HW
  // cluster split-barrier (s_barrier_signal/wait -3) instead.
  lstm_kernel<<<32, 256, 0, stream>>>(featbf, wifb, whfb, wirb, whrb, bif, bhf, bir, bhr, hsf, hsb,
                                      zrow, bars, /*use_cluster=*/0);
  eenc_kernel<<<1000, 256, 0, stream>>>(hsf, hsb, wewbf, web, eenc);

  dec_init_kernel<<<(BSZ * HDEC + 255) / 256, 256, 0, stream>>>(c0, ctxbf, ctxf, hdA, cd);
  for (int s = 0; s < 10; ++s) {
    unsigned short* hdin  = (s & 1) ? hdB : hdA;
    unsigned short* hdout = (s & 1) ? hdA : hdB;
    dec_gate_kernel<<<32, 256, 0, stream>>>(ctxbf, hdin, dwib, dwhb, dbi, dbh, cd, hdout);
    dec_d_kernel<<<16, 256, 0, stream>>>(hdout, wdwb, wdb, dbuf);
    attn_score_kernel<<<(BSZ * LSEQ) / 8, 256, 0, stream>>>(eenc, dbuf, attnv, abuf);
    attn_ctx_kernel<<<BSZ, 256, 0, stream>>>(abuf, hsf, hsb, ctxf, ctxbf);
  }
  head_kernel<<<BSZ, 256, 0, stream>>>(ctxf, fcw, fcb, ow, ob, (float*)d_out);
}
